// KlearSparseMoeBlock_43963285242589
// MI455X (gfx1250) — matrix operand order repaired and checked by reading, not compiled, not run on hardware
//
#include <hip/hip_runtime.h>

// Problem dims (fixed by the reference setup)
#define S_TOK   2048
#define DMODEL  2048
#define NEXP    32
#define HEXP    512
#define HSHARED 1024
#define TOPK    4
#define NWAVES  4   // waves per GEMM block

typedef __attribute__((ext_vector_type(16))) __bf16 v16bf;
typedef __attribute__((ext_vector_type(8)))  __bf16 v8bf;
typedef __attribute__((ext_vector_type(8)))  float  v8f;

// Native f32 -> bf16 convert (clang lowers to v_cvt_pk_bf16_f32-class ops).
__device__ __forceinline__ __bf16 f2bf(float f) { return (__bf16)f; }

__device__ __forceinline__ void pack4(v16bf& frag, int base, float4 f) {
  frag[base + 0] = f2bf(f.x);
  frag[base + 1] = f2bf(f.y);
  frag[base + 2] = f2bf(f.z);
  frag[base + 3] = f2bf(f.w);
}

// Fast sigmoid: raw v_rcp_f32 instead of IEEE divide expansion.
__device__ __forceinline__ float fast_sigmoid(float v) {
  return __builtin_amdgcn_rcpf(1.f + __expf(-v));
}

// ---------------------------------------------------------------------------
// Router: one wave per token, lane = expert. sigmoid -> bias-adjusted top-4
// (ties -> lowest index, matching lax.top_k), normalized UNbiased scores,
// scatter (token, score) into per-expert lists via atomics.
// ---------------------------------------------------------------------------
__global__ void router_kernel(const float* __restrict__ x,
                              const float* __restrict__ gate_w,
                              const float* __restrict__ expert_bias,
                              int* __restrict__ counts,
                              int* __restrict__ tlist,
                              float* __restrict__ slist) {
  const int wave = threadIdx.x >> 5;
  const int lane = threadIdx.x & 31;
  const int t = blockIdx.x * (blockDim.x >> 5) + wave;

  const float* xr = x + (size_t)t * DMODEL;
  const float* wr = gate_w + (size_t)lane * DMODEL;
  float acc = 0.f;
  for (int d = 0; d < DMODEL; d += 4) {
    const float4 xv = *(const float4*)(xr + d);
    const float4 wv = *(const float4*)(wr + d);
    acc += xv.x * wv.x + xv.y * wv.y + xv.z * wv.z + xv.w * wv.w;
  }
  const float r = fast_sigmoid(acc);
  float cur = r + expert_bias[lane];

  int inds[TOPK];
  float scs[TOPK];
  float ssum = 0.f;
  for (int k = 0; k < TOPK; ++k) {
    float bv = cur;
    int bi = lane;
    for (int off = 16; off; off >>= 1) {
      float ov = __shfl_xor(bv, off);
      int oi = __shfl_xor(bi, off);
      if (ov > bv || (ov == bv && oi < bi)) { bv = ov; bi = oi; }
    }
    inds[k] = bi;
    float rw = __shfl(r, bi);
    scs[k] = rw;
    ssum += rw;
    if (lane == bi) cur = -1e30f;
  }
  if (lane == 0) {
    const float inv = __builtin_amdgcn_rcpf(ssum + 1e-20f);
    for (int k = 0; k < TOPK; ++k) {
      const int e = inds[k];
      const int pos = atomicAdd(&counts[e], 1);
      tlist[e * S_TOK + pos] = t;
      slist[e * S_TOK + pos] = scs[k] * inv;
    }
  }
}

// ---------------------------------------------------------------------------
// 2-way mixing coefficient: wave per token, strided partials + shfl reduce.
// ---------------------------------------------------------------------------
__global__ void coef_kernel(const float* __restrict__ x,
                            const float* __restrict__ coef_w,
                            const float* __restrict__ coef_b,
                            float* __restrict__ coef) {
  const int wave = threadIdx.x >> 5;
  const int lane = threadIdx.x & 31;
  const int t = blockIdx.x * (blockDim.x >> 5) + wave;
  const float* xr = x + (size_t)t * DMODEL;
  float a0 = 0.f, a1 = 0.f;
  for (int d = lane; d < DMODEL; d += 32) {
    const float xv = xr[d];
    a0 += xv * coef_w[d];
    a1 += xv * coef_w[DMODEL + d];
  }
  for (int off = 16; off; off >>= 1) {
    a0 += __shfl_xor(a0, off);
    a1 += __shfl_xor(a1, off);
  }
  if (lane == 0) {
    a0 += coef_b[0];
    a1 += coef_b[1];
    const float m = fmaxf(a0, a1);
    const float e0 = __expf(a0 - m), e1 = __expf(a1 - m);
    const float inv = __builtin_amdgcn_rcpf(e0 + e1);
    coef[t * 2 + 0] = e0 * inv;
    coef[t * 2 + 1] = e1 * inv;
  }
}

// ---------------------------------------------------------------------------
// Fused SwiGLU expert GEMM: one block (NWAVES waves) per 16-token tile.
//   GATHER=true : tokens gathered from per-expert lists, act scaled by routing
//                 score, results atomically accumulated into `out`.
//   GATHER=false: shared expert, identity tokens, plain stores.
//
// WMMA bf16 16x16x32 fragment layouts used by this code (contiguous k, not the ISA table order):
//   A (16x32): lane l (half=l>>4, l16=l&15) holds row M=l16,
//              elem j -> K = half*16 + j
//   B (32x16): lane l holds column N=l16, elem j -> K = half*16 + j
//   C/D      : lane l holds column N=l16, elem r -> row M = r + half*8
// Waves cooperatively stage x-fragments, then split the n-tile loops.
// ---------------------------------------------------------------------------
template <int H, bool GATHER>
__global__ void glu_gemm_kernel(const float* __restrict__ x,
                                const float* __restrict__ wg_,
                                const float* __restrict__ wu_,
                                const float* __restrict__ wd_,
                                const int* __restrict__ counts,
                                const int* __restrict__ tlist,
                                const float* __restrict__ slist,
                                float* __restrict__ out) {
  constexpr int CH1 = DMODEL / 32;  // 64 k-chunks over D
  constexpr int CH2 = H / 32;       // k-chunks over H
  constexpr int NT1 = H / 16;       // n-tiles for gate/up
  constexpr int NT2 = DMODEL / 16;  // 128 n-tiles for down

  __shared__ __attribute__((aligned(32))) __bf16 ldsx[CH1 * 32 * 16];  // 64 KB: prepacked A frags
  __shared__ __attribute__((aligned(16))) __bf16 actlds[16 * H];       // 16/32 KB: activation tile

  const int wv = threadIdx.x >> 5;   // wave id within block
  const int lane = threadIdx.x & 31;
  const int l16 = lane & 15;
  const int half = lane >> 4;
  const int tile = blockIdx.x;
  const int e = GATHER ? (int)blockIdx.y : 0;

  int cnt = S_TOK;
  if constexpr (GATHER) {
    cnt = counts[e];
    if (tile * 16 >= cnt) return;  // uniform across block
  }

  const float* wgp = wg_ + (GATHER ? (size_t)e * DMODEL * H : (size_t)0);
  const float* wup = wu_ + (GATHER ? (size_t)e * DMODEL * H : (size_t)0);
  const float* wdp = wd_ + (GATHER ? (size_t)e * H * DMODEL : (size_t)0);

  // Per-lane output rows (m = r + half*8): token id + routing score.
  int tk[8];
  float sc[8];
#pragma unroll
  for (int r = 0; r < 8; ++r) {
    const int gi = tile * 16 + r + half * 8;
    if constexpr (GATHER) {
      const bool valid = gi < cnt;
      tk[r] = valid ? tlist[e * S_TOK + gi] : 0;
      sc[r] = valid ? slist[e * S_TOK + gi] : 0.f;  // padded rows contribute 0
    } else {
      tk[r] = gi;
      sc[r] = 1.f;
    }
  }
  // A-fragment row for this lane (row M = l16).
  int rowTok;
  {
    const int gi = tile * 16 + l16;
    if constexpr (GATHER) rowTok = (gi < cnt) ? tlist[e * S_TOK + gi] : 0;
    else rowTok = gi;
  }

  // ---- Cooperatively stage x tile as prepacked bf16 WMMA-A frags in LDS ----
  // Fragment content depends only on (chunk, lane-pattern), so waves split
  // the chunk range.
  const float* xrow = x + (size_t)rowTok * DMODEL;
  for (int c = wv; c < CH1; c += NWAVES) {
    const int k0 = c * 32;
    const float* p0 = xrow + k0 + half * 16;
    const float* p1 = xrow + k0 + half * 16 + 8;
    const float4 f0 = *(const float4*)(p0);
    const float4 f1 = *(const float4*)(p0 + 4);
    const float4 f2 = *(const float4*)(p1);
    const float4 f3 = *(const float4*)(p1 + 4);
    v16bf frag;
    pack4(frag, 0, f0);
    pack4(frag, 4, f1);
    pack4(frag, 8, f2);
    pack4(frag, 12, f3);
    *(v16bf*)&ldsx[(c * 32 + lane) * 16] = frag;
  }
  __syncthreads();

  // ---- Phase 1: gate & up GEMMs + SwiGLU * score -> actlds (waves split nt)
  for (int nt = wv; nt < NT1; nt += NWAVES) {
    const int h0 = nt * 16;
    v8f accg = {};
    v8f accu = {};
    for (int c = 0; c < CH1; ++c) {
      const v16bf a = *(const v16bf*)&ldsx[(c * 32 + lane) * 16];
      const float* pg = wgp + (size_t)(c * 32 + half * 16) * H + h0 + l16;
      const float* pu = wup + (size_t)(c * 32 + half * 16) * H + h0 + l16;
      v16bf bg, bu;
#pragma unroll
      for (int j = 0; j < 16; ++j) {
        bg[j] = f2bf(pg[(size_t)j * H]);  // half-wave coalesced (16 consecutive h)
        bu[j] = f2bf(pu[(size_t)j * H]);
      }
      accg = __builtin_amdgcn_wmma_f32_16x16x32_bf16(false, a, false, bg, (short)0, accg, false, false);
      accu = __builtin_amdgcn_wmma_f32_16x16x32_bf16(false, a, false, bu, (short)0, accu, false, false);
    }
#pragma unroll
    for (int r = 0; r < 8; ++r) {
      const int m = r + half * 8;
      const float g = accg[r];
      const float act = g * fast_sigmoid(g) * accu[r] * sc[r];
      actlds[m * H + h0 + l16] = f2bf(act);
    }
  }
  __syncthreads();

  // ---- Phase 2: down projection, accumulate into y (waves split nt) ----
  for (int nt = wv; nt < NT2; nt += NWAVES) {
    const int d0 = nt * 16;
    v8f acc = {};
    for (int c = 0; c < CH2; ++c) {
      const int k0 = c * 32;
      const v8bf s0 = *(const v8bf*)&actlds[l16 * H + k0 + half * 16];
      const v8bf s1 = *(const v8bf*)&actlds[l16 * H + k0 + half * 16 + 8];
      const v16bf a = __builtin_shufflevector(s0, s1, 0, 1, 2, 3, 4, 5, 6, 7,
                                              8, 9, 10, 11, 12, 13, 14, 15);
      const float* pd = wdp + (size_t)(k0 + half * 16) * DMODEL + d0 + l16;
      v16bf b;
#pragma unroll
      for (int j = 0; j < 16; ++j) b[j] = f2bf(pd[(size_t)j * DMODEL]);
      acc = __builtin_amdgcn_wmma_f32_16x16x32_bf16(false, a, false, b, (short)0, acc, false, false);
    }
#pragma unroll
    for (int r = 0; r < 8; ++r) {
      if constexpr (GATHER) {
        atomicAdd(&out[(size_t)tk[r] * DMODEL + d0 + l16], acc[r]);
      } else {
        out[(size_t)tk[r] * DMODEL + d0 + l16] = acc[r];
      }
    }
  }
}

// ---------------------------------------------------------------------------
// out = y_experts * coef0 + shared * coef1
// ---------------------------------------------------------------------------
__global__ void finalize_kernel(const float* __restrict__ y,
                                const float* __restrict__ sh,
                                const float* __restrict__ coef,
                                float* __restrict__ out) {
  const size_t i = (size_t)blockIdx.x * blockDim.x + threadIdx.x;
  const int t = (int)(i / DMODEL);
  out[i] = y[i] * coef[t * 2 + 0] + sh[i] * coef[t * 2 + 1];
}

extern "C" void kernel_launch(void* const* d_in, const int* in_sizes, int n_in,
                              void* d_out, int out_size, void* d_ws, size_t ws_size,
                              hipStream_t stream) {
  const float* x           = (const float*)d_in[0];
  const float* gate_w      = (const float*)d_in[1];
  const float* expert_bias = (const float*)d_in[2];
  const float* wg          = (const float*)d_in[3];
  const float* wu          = (const float*)d_in[4];
  const float* wd          = (const float*)d_in[5];
  const float* sg          = (const float*)d_in[6];
  const float* su          = (const float*)d_in[7];
  const float* sd          = (const float*)d_in[8];
  const float* coef_w      = (const float*)d_in[9];
  const float* coef_b      = (const float*)d_in[10];
  // d_in[11] = top_k (fixed to 4 here)
  (void)in_sizes; (void)n_in; (void)out_size; (void)ws_size;

  float* out = (float*)d_out;

  // Workspace carve-up (~34.3 MB)
  char* ws = (char*)d_ws;
  float* y = (float*)ws;            ws += sizeof(float) * (size_t)S_TOK * DMODEL;
  float* shbuf = (float*)ws;        ws += sizeof(float) * (size_t)S_TOK * DMODEL;
  float* coef = (float*)ws;         ws += sizeof(float) * (size_t)S_TOK * 2;
  float* slist = (float*)ws;        ws += sizeof(float) * (size_t)NEXP * S_TOK;
  int* counts = (int*)ws;           ws += sizeof(int) * 64;
  int* tlist = (int*)ws;

  hipMemsetAsync(counts, 0, sizeof(int) * 64, stream);
  hipMemsetAsync(y, 0, sizeof(float) * (size_t)S_TOK * DMODEL, stream);

  router_kernel<<<S_TOK / 8, 256, 0, stream>>>(x, gate_w, expert_bias, counts, tlist, slist);
  coef_kernel<<<S_TOK / 8, 256, 0, stream>>>(x, coef_w, coef_b, coef);

  // Top-4 experts: grid covers worst case (all tokens -> one expert)
  glu_gemm_kernel<HEXP, true><<<dim3(S_TOK / 16, NEXP), 32 * NWAVES, 0, stream>>>(
      x, wg, wu, wd, counts, tlist, slist, y);

  // Shared expert
  glu_gemm_kernel<HSHARED, false><<<dim3(S_TOK / 16, 1), 32 * NWAVES, 0, stream>>>(
      x, sg, su, sd, nullptr, nullptr, nullptr, shbuf);

  finalize_kernel<<<((size_t)S_TOK * DMODEL) / 256, 256, 0, stream>>>(y, shbuf, coef, out);
}
